// Semantic_Caption_Model_LSTMCell_37632503447981
// MI455X (gfx1250) — matrix operand order repaired and checked by reading, not compiled, not run on hardware
//
#include <hip/hip_runtime.h>
#include <hip/hip_bf16.h>
#include <math.h>

// ---------------------------------------------------------------------------
// Types for CDNA5 WMMA (wave32): v_wmma_f32_16x16x32_bf16
// ---------------------------------------------------------------------------
typedef __bf16 bf16_t;
typedef bf16_t v16bf __attribute__((ext_vector_type(16)));
typedef float  v8f   __attribute__((ext_vector_type(8)));

#define TN 64    // cols per block
#define TK 32    // K step (matches 16x16x32 WMMA)
#define LDSP 40  // padded LDS row stride in ushorts (80B, 16B aligned)

struct alignas(32) FragBits { unsigned int u[8]; };

__device__ __forceinline__ unsigned int rnd_bf16_hi(float f) {
  unsigned int u = __float_as_uint(f);
  return u + 0x7FFFu + ((u >> 16) & 1u);   // RNE in bits [31:16]
}
// pack two fp32 -> two bf16 in one dword via v_perm_b32
__device__ __forceinline__ unsigned int pk(float lo, float hi) {
  return __builtin_amdgcn_perm(rnd_bf16_hi(hi), rnd_bf16_hi(lo), 0x07060302u);
}
__device__ __forceinline__ v16bf ld_frag(const unsigned short* p0, const unsigned short* p1) {
  uint4 a = *(const uint4*)p0;
  uint4 b = *(const uint4*)p1;
  FragBits fb;
  fb.u[0] = a.x; fb.u[1] = a.y; fb.u[2] = a.z; fb.u[3] = a.w;
  fb.u[4] = b.x; fb.u[5] = b.y; fb.u[6] = b.z; fb.u[7] = b.w;
  return __builtin_bit_cast(v16bf, fb);
}
__device__ __forceinline__ float sigmoidf_(float x) { return 1.0f / (1.0f + expf(-x)); }
__device__ __forceinline__ int imin_(int a, int b) { return a < b ? a : b; }

// ---------------------------------------------------------------------------
// NT GEMM with optional fused second operand pair:
//   C[M,N] = A[M,K(lda)] @ W[N,K]^T  (+ A2[M,K2(lda2)] @ W2[N,K2]^T)
//            (+bias[N]) (+bias2[N]) (+resid[M,N])
// fp32 in/out, bf16 WMMA math, fp32 accumulate, double-buffered LDS.
// BM=128: 8 waves = 8 row-blocks x 1 col-group (4 accs each)
// BM=64 : 8 waves = 4 row-blocks x 2 col-groups (2 accs each) -- for M<=64
// ---------------------------------------------------------------------------
template<int BM>
__global__ __launch_bounds__(256)
void gemm_nt_wmma(const float* __restrict__ A,  int lda,  const float* __restrict__ W,
                  const float* __restrict__ A2, int lda2, const float* __restrict__ W2,
                  const float* __restrict__ bias, const float* __restrict__ bias2,
                  const float* __restrict__ resid,
                  float* __restrict__ C,
                  int M, int N, int K, int K2)
{
  constexpr int RB  = BM / 16;   // row blocks
  constexpr int CG  = 8 / RB;    // column groups
  constexpr int JJN = 4 / CG;    // 16-col subtiles per wave
  constexpr int AF  = (BM == 128) ? 4 : 2;  // float4 regs for A staging

  __shared__ unsigned short Ash[2][BM][LDSP];
  __shared__ unsigned short Bsh[2][TN][LDSP];

  const int tid  = threadIdx.x;
  const int wave = tid >> 5;
  const int lane = tid & 31;
  const int m0 = blockIdx.y * BM;
  const int n0 = blockIdx.x * TN;

  // staging coords (row clamped -> no branches, padded rows/cols are
  // harmless: they only feed C rows/cols that are never stored)
  const int arow  = (BM == 128) ? (tid >> 1) : (tid >> 2);
  const int akoff = (BM == 128) ? ((tid & 1) * 16) : ((tid & 3) * 8);
  const int brow  = tid >> 2;
  const int bkoff = (tid & 3) * 8;
  const int am = imin_(m0 + arow, M - 1);
  const int bn = imin_(n0 + brow, N - 1);

  // fragment coords (ISA 7.12.2 layouts)
  const int lm   = lane & 15;
  const int hi   = lane >> 4;
  const int a_kb = hi * 8;
  const int b_kb = hi * 8;
  const int wrow = wave % RB;
  const int wcol = wave / RB;

  v8f acc[JJN] = {};
  int pp = 0;

  auto run_phase = [&](const float* Ap, int ldap, const float* Wp, int Kp) {
    const float* aptr = Ap + (size_t)am * ldap + akoff;
    const float* bptr = Wp + (size_t)bn * Kp + bkoff;
    float4 fa[AF];
    float4 fb[2];

    auto load_g = [&](int k0) {
      const float4* as = (const float4*)(aptr + k0);
#pragma unroll
      for (int i = 0; i < AF; ++i) fa[i] = as[i];
      const float4* bs = (const float4*)(bptr + k0);
      fb[0] = bs[0]; fb[1] = bs[1];
      if (k0 + 2 * TK < Kp)  // global_prefetch_b8 two tiles ahead
        __builtin_prefetch(bptr + k0 + 2 * TK, 0, 3);
    };
    auto store_l = [&](int p) {
      *(uint4*)&Ash[p][arow][akoff] =
          make_uint4(pk(fa[0].x, fa[0].y), pk(fa[0].z, fa[0].w),
                     pk(fa[1].x, fa[1].y), pk(fa[1].z, fa[1].w));
      if constexpr (BM == 128) {
        *(uint4*)&Ash[p][arow][akoff + 8] =
            make_uint4(pk(fa[2].x, fa[2].y), pk(fa[2].z, fa[2].w),
                       pk(fa[3].x, fa[3].y), pk(fa[3].z, fa[3].w));
      }
      *(uint4*)&Bsh[p][brow][bkoff] =
          make_uint4(pk(fb[0].x, fb[0].y), pk(fb[0].z, fb[0].w),
                     pk(fb[1].x, fb[1].y), pk(fb[1].z, fb[1].w));
    };

    load_g(0);
    store_l(pp);
    for (int k0 = 0; k0 < Kp; k0 += TK) {
      __syncthreads();
      const bool more = (k0 + TK) < Kp;
      if (more) load_g(k0 + TK);

      const unsigned short* ap = &Ash[pp][wrow * 16 + lm][0];
      v16bf af = ld_frag(ap + a_kb, ap + a_kb + 16);
#pragma unroll
      for (int jj = 0; jj < JJN; ++jj) {
        const unsigned short* bp = &Bsh[pp][wcol * (JJN * 16) + jj * 16 + lm][b_kb];
        v16bf bf = ld_frag(bp, bp + 16);
        acc[jj] = __builtin_amdgcn_wmma_f32_16x16x32_bf16(
            false, af, false, bf, (short)0, acc[jj], false, false);
      }
      __syncthreads();
      if (more) store_l(pp ^ 1);
      pp ^= 1;
    }
  };

  run_phase(A, lda, W, K);
  if (A2) run_phase(A2, lda2, W2, K2);

  // ---- epilogue ----
  const int mbase = m0 + wrow * 16 + hi * 8;
#pragma unroll
  for (int jj = 0; jj < JJN; ++jj) {
    const int n = n0 + wcol * (JJN * 16) + jj * 16 + lm;
    if (n >= N) continue;
    float bv = (bias ? bias[n] : 0.0f) + (bias2 ? bias2[n] : 0.0f);
#pragma unroll
    for (int e = 0; e < 8; ++e) {
      const int m = mbase + e;
      if (m < M) {
        const size_t idx = (size_t)m * N + n;
        const float r = resid ? resid[idx] : 0.0f;
        C[idx] = acc[jj][e] + bv + r;
      }
    }
  }
}

// ---------------------------------------------------------------------------
// Pointwise / gather / attention kernels (fp32, tiny fraction of runtime)
// ---------------------------------------------------------------------------
__global__ void build_input_emb_k(const float* __restrict__ first_inputs,
                                  const float* __restrict__ wte,
                                  const int* __restrict__ cap,
                                  float* __restrict__ emb)
{
  int i = blockIdx.x * blockDim.x + threadIdx.x;
  if (i >= 64 * 20 * 768) return;
  int e = i % 768, t = (i / 768) % 20, b = i / (768 * 20);
  float v = (t == 0) ? first_inputs[b * 768 + e]
                     : wte[(size_t)cap[b * 20 + (t - 1)] * 768 + e];
  emb[i] = v;
}

__global__ void gather_attr_k(const float* __restrict__ wte,
                              const int* __restrict__ ids,
                              float* __restrict__ attr,
                              float* __restrict__ s_attr)
{
  int i = blockIdx.x * blockDim.x + threadIdx.x;
  if (i >= 64 * 5 * 768) return;
  int e = i % 768, ba = i / 768;
  float v = wte[(size_t)ids[ba] * 768 + e];
  attr[i] = v;
  s_attr[i] = sigmoidf_(v);
}

__global__ __launch_bounds__(256)
void attention_k(const float* __restrict__ q,      // [B*T,768]
                 const float* __restrict__ av,     // [B,5,768]
                 float* __restrict__ ctx,          // [B*T,768]
                 int T)
{
  const int bt = blockIdx.x;
  const int b = bt / T;
  const int tid = threadIdx.x;
  __shared__ float red[256];
  __shared__ float wsh[5];

  float part[5] = {0.f, 0.f, 0.f, 0.f, 0.f};
  const float* qrow = q + (size_t)bt * 768;
  for (int e = tid; e < 768; e += 256) {
    float qv = qrow[e];
#pragma unroll
    for (int a = 0; a < 5; ++a)
      part[a] += qv * av[((size_t)b * 5 + a) * 768 + e];
  }
#pragma unroll
  for (int a = 0; a < 5; ++a) {
    red[tid] = part[a];
    __syncthreads();
    for (int s = 128; s > 0; s >>= 1) {
      if (tid < s) red[tid] += red[tid + s];
      __syncthreads();
    }
    if (tid == 0) wsh[a] = red[0];
    __syncthreads();
  }
  if (tid == 0) {
    float mx = wsh[0];
    for (int a = 1; a < 5; ++a) mx = fmaxf(mx, wsh[a]);
    float sum = 0.f;
    for (int a = 0; a < 5; ++a) { wsh[a] = expf(wsh[a] - mx); sum += wsh[a]; }
    float inv = 1.0f / sum;
    for (int a = 0; a < 5; ++a) wsh[a] *= inv;
  }
  __syncthreads();
  for (int e = tid; e < 768; e += 256) {
    float acc = 0.f;
#pragma unroll
    for (int a = 0; a < 5; ++a)
      acc += wsh[a] * av[((size_t)b * 5 + a) * 768 + e];
    ctx[(size_t)bt * 768 + e] = acc;
  }
}

__global__ void lstm_pointwise_k(const float* __restrict__ gates,
                                 float* __restrict__ h,
                                 float* __restrict__ c,
                                 float* __restrict__ hs,
                                 int t)
{
  int i = blockIdx.x * blockDim.x + threadIdx.x;
  if (i >= 64 * 1024) return;
  int d = i % 1024, b = i / 1024;
  const float* g = gates + (size_t)b * 4096;
  float ig = sigmoidf_(g[d]);
  float fg = sigmoidf_(g[1024 + d]);
  float gg = tanhf(g[2048 + d]);
  float og = sigmoidf_(g[3072 + d]);
  float c2 = fg * c[i] + ig * gg;
  float h2 = og * tanhf(c2);
  c[i] = c2;
  h[i] = h2;
  hs[((size_t)b * 20 + t) * 1024 + d] = h2;
}

// ---------------------------------------------------------------------------
// Orchestration
// ---------------------------------------------------------------------------
extern "C" void kernel_launch(void* const* d_in, const int* in_sizes, int n_in,
                              void* d_out, int out_size, void* d_ws, size_t ws_size,
                              hipStream_t stream)
{
  (void)in_sizes; (void)n_in; (void)out_size; (void)ws_size;

  const int*   cap       = (const int*)  d_in[0];
  const float* fm        = (const float*)d_in[1];   // [64, 32768]
  const int*   attr_ids  = (const int*)  d_in[2];
  const float* wte       = (const float*)d_in[3];
  const float* fc_w      = (const float*)d_in[4],  *fc_b       = (const float*)d_in[5];
  const float* hh_w      = (const float*)d_in[6],  *hh_b       = (const float*)d_in[7];
  const float* cc_w      = (const float*)d_in[8],  *cc_b       = (const float*)d_in[9];
  const float* lm_w      = (const float*)d_in[10], *lm_b       = (const float*)d_in[11];
  const float* iaw_w     = (const float*)d_in[12], *iaw_b      = (const float*)d_in[13];
  const float* iau_w     = (const float*)d_in[14], *iau_b      = (const float*)d_in[15];
  const float* indiag_w  = (const float*)d_in[16], *indiag_b   = (const float*)d_in[17];
  const float* indomain_w= (const float*)d_in[18], *indomain_b = (const float*)d_in[19];
  const float* oav_w     = (const float*)d_in[20], *oav_b      = (const float*)d_in[21];
  const float* outdiag_w = (const float*)d_in[22], *outdiag_b  = (const float*)d_in[23];
  const float* outdomain_w=(const float*)d_in[24], *outdomain_b= (const float*)d_in[25];
  const float* w_ih      = (const float*)d_in[26], *b_ih       = (const float*)d_in[27];
  const float* w_hh      = (const float*)d_in[28], *b_hh       = (const float*)d_in[29];

  float* ws = (float*)d_ws;
  size_t off = 0;
  auto alloc = [&](size_t n) { float* p = ws + off; off += n; return p; };
  float* first_inputs = alloc(64 * 768);
  float* input_emb    = alloc(1280 * 768);
  float* mi_init      = alloc(64 * 1024);
  float* attr         = alloc(64 * 5 * 768);
  float* s_attr       = alloc(64 * 5 * 768);
  float* m1           = alloc(1280 * 768);
  float* ctx          = alloc(1280 * 768);
  float* g1           = alloc(1280 * 768);
  float* model_inputs = alloc(1280 * 1024);
  float* h            = alloc(64 * 1024);
  float* c            = alloc(64 * 1024);
  float* gates        = alloc(64 * 4096);
  float* hs           = alloc(1280 * 1024);
  float* g2           = alloc(1280 * 1024);
  float* outb         = alloc(1280 * 1024);

  auto gemm = [&](const float* A, int lda, const float* Wm,
                  const float* bias, const float* resid, float* Cp,
                  int M, int N, int K) {
    if (M <= 64) {
      dim3 grid((N + TN - 1) / TN, (M + 63) / 64);
      gemm_nt_wmma<64><<<grid, dim3(256), 0, stream>>>(
          A, lda, Wm, nullptr, 0, nullptr, bias, nullptr, resid, Cp, M, N, K, 0);
    } else {
      dim3 grid((N + TN - 1) / TN, (M + 127) / 128);
      gemm_nt_wmma<128><<<grid, dim3(256), 0, stream>>>(
          A, lda, Wm, nullptr, 0, nullptr, bias, nullptr, resid, Cp, M, N, K, 0);
    }
  };

  // 1) first_inputs = fm @ fc_w^T + fc_b            [64,768]  K=32768
  gemm(fm, 32768, fc_w, fc_b, nullptr, first_inputs, 64, 768, 32768);
  // 2) input_emb = concat(first_inputs, wte[cap[:, :-1]])
  build_input_emb_k<<<(64 * 20 * 768 + 255) / 256, 256, 0, stream>>>(first_inputs, wte, cap, input_emb);
  // 3) mi_init = first_inputs @ iaw_w^T + iaw_b     [64,1024]
  gemm(first_inputs, 768, iaw_w, iaw_b, nullptr, mi_init, 64, 1024, 768);
  // 4) attr = wte[attributes]; s_attr = sigmoid(attr)
  gather_attr_k<<<(64 * 5 * 768 + 255) / 256, 256, 0, stream>>>(wte, attr_ids, attr, s_attr);
  // 5) m1 = input_emb @ iau_w^T + iau_b             [1280,768]
  gemm(input_emb, 768, iau_w, iau_b, nullptr, m1, 1280, 768, 768);
  // 6) input attention -> ctx
  attention_k<<<1280, 256, 0, stream>>>(m1, attr, ctx, 20);
  // 7) g1 = input_emb + ctx @ indiag_w^T + indiag_b [1280,768]
  gemm(ctx, 768, indiag_w, indiag_b, input_emb, g1, 1280, 768, 768);
  // 8) model_inputs = g1 @ indomain_w^T + b         [1280,1024]
  gemm(g1, 768, indomain_w, indomain_b, nullptr, model_inputs, 1280, 1024, 768);
  // 9) hidden / cell projections                    [64,1024] K=32768
  gemm(fm, 32768, hh_w, hh_b, nullptr, h, 64, 1024, 32768);
  gemm(fm, 32768, cc_w, cc_b, nullptr, c, 64, 1024, 32768);
  // 10) LSTM: fused gates = x@w_ih^T + h@w_hh^T + b_ih + b_hh, then pointwise
  for (int t = 0; t < 20; ++t) {
    const float* x   = (t == 0) ? mi_init : (model_inputs + (size_t)t * 1024);
    const int    lda = (t == 0) ? 1024 : (20 * 1024);
    dim3 grid((4096 + TN - 1) / TN, 1);
    gemm_nt_wmma<64><<<grid, dim3(256), 0, stream>>>(
        x, lda, w_ih, h, 1024, w_hh, b_ih, b_hh, nullptr, gates, 64, 4096, 1024, 1024);
    lstm_pointwise_k<<<(64 * 1024 + 255) / 256, 256, 0, stream>>>(gates, h, c, hs, t);
  }
  // 11) m1o = hs @ oav_w^T + oav_b                  [1280,768]
  gemm(hs, 1024, oav_w, oav_b, nullptr, m1, 1280, 768, 1024);
  // 12) output attention (sigmoid-gated attrs)
  attention_k<<<1280, 256, 0, stream>>>(m1, s_attr, ctx, 20);
  // 13) g2 = hs + ctx_o @ outdiag_w^T + b           [1280,1024]
  gemm(ctx, 768, outdiag_w, outdiag_b, hs, g2, 1280, 1024, 768);
  // 14) out = g2 @ outdomain_w^T + b                [1280,1024]
  gemm(g2, 1024, outdomain_w, outdomain_b, nullptr, outb, 1280, 1024, 1024);
  // 15) logits = out @ lm_w^T + lm_b                [1280,50260]
  gemm(outb, 1024, lm_w, lm_b, nullptr, (float*)d_out, 1280, 50260, 1024);
}
